// UpdatePaiNN_66554813219143
// MI455X (gfx1250) — matrix operand order repaired and checked by reading, not compiled, not run on hardware
//
#include <hip/hip_runtime.h>
#include <hip/hip_bf16.h>

// ---------------------------------------------------------------------------
// PaiNN update block, fused, bf16 WMMA + TDM (gfx1250 / CDNA5, wave32)
//   N = 200000 rows, F = 128 features
//   outputs: deltasu [N,F] f32, then deltavu [N,F,3] f32 (concatenated)
// ---------------------------------------------------------------------------

#define NROWS  200000
#define FDIM   128
#define TILE_N 64     // rows per workgroup; 200000 = 64 * 3125 exactly

typedef __bf16        bf16x8  __attribute__((ext_vector_type(8)));
typedef __bf16        bf16x16 __attribute__((ext_vector_type(16)));
typedef float         f32x8   __attribute__((ext_vector_type(8)));
typedef unsigned int  u32x4   __attribute__((ext_vector_type(4)));
typedef int           i32x4   __attribute__((ext_vector_type(4)));
typedef int           i32x8   __attribute__((ext_vector_type(8)));

#if defined(__has_builtin)
#if __has_builtin(__builtin_amdgcn_tensor_load_to_lds) && \
    __has_builtin(__builtin_amdgcn_tensor_store_from_lds) && \
    __has_builtin(__builtin_amdgcn_s_wait_tensorcnt)
#define HAVE_TDM 1
#endif
#endif
#ifndef HAVE_TDM
#define HAVE_TDM 0
#endif

// ---- WMMA wrapper: D(f32 16x16) = A(bf16 16x32) * B(bf16 32x16) + C --------
__device__ __forceinline__ f32x8 wmma_bf16(bf16x16 a, bf16x16 b, f32x8 c) {
    return __builtin_amdgcn_wmma_f32_16x16x32_bf16(
        /*neg_a=*/false, a, /*neg_b=*/false, b,
        /*c_mod=*/(short)0, c, /*reuse_a=*/false, /*reuse_b=*/false);
}

// ---- A-fragment (16x32 bf16), ISA 7.12.2 layout ----------------------------
__device__ __forceinline__ bf16x16
ld_frag_a(const __hip_bfloat16* base, int ld, int m0, int k0, int lane) {
    const int row  = lane & 15;
    const int half = lane >> 4;
    const __bf16* p =
        reinterpret_cast<const __bf16*>(base) + (m0 + row) * ld + k0 + half * 8;
    bf16x8 lo = *reinterpret_cast<const bf16x8*>(p);
    bf16x8 hi = *reinterpret_cast<const bf16x8*>(p + 16);
    return __builtin_shufflevector(lo, hi, 0, 1, 2, 3, 4, 5, 6, 7,
                                           8, 9, 10, 11, 12, 13, 14, 15);
}

// ---- B-fragment (32x16 bf16) from PRE-TRANSPOSED weight Bt[g][k] -----------
__device__ __forceinline__ bf16x16
ld_frag_b(const __hip_bfloat16* Bt, int ld, int n0, int k0, int lane) {
    const int col = lane & 15;
    const int sel = lane >> 4;
    const __bf16* p =
        reinterpret_cast<const __bf16*>(Bt) + (n0 + col) * ld + k0 + sel * 8;
    bf16x8 lo = *reinterpret_cast<const bf16x8*>(p);
    bf16x8 hi = *reinterpret_cast<const bf16x8*>(p + 16);
    return __builtin_shufflevector(lo, hi, 0, 1, 2, 3, 4, 5, 6, 7,
                                           8, 9, 10, 11, 12, 13, 14, 15);
}

// ---- TDM descriptor: 1-D contiguous transfer, data_size = 8 bytes ----------
// D# per CDNA5 ISA ch.8: group0 = {count, lds_addr, global_addr(57b), type=2},
// group1 = {data_size, tensor_dim0/1, tile_dim0/1, dim0 stride}. units8 must
// fit in 16 bits (tile_dim0).
struct TdmDesc { u32x4 g0; i32x8 g1; };

__device__ __forceinline__ TdmDesc tdm_desc_1d(unsigned int lds_byte_off,
                                               const void* gptr,
                                               unsigned int units8) {
    TdmDesc d;
    const unsigned long long ga = (unsigned long long)(uintptr_t)gptr;
    d.g0[0] = 1u;                                   // count=1 (valid user D#)
    d.g0[1] = lds_byte_off;                         // lds_addr
    d.g0[2] = (unsigned int)ga;                     // global_addr[31:0]
    d.g0[3] = (unsigned int)((ga >> 32) & 0x01FFFFFFu) | (2u << 30); // [56:32]|type=2
    d.g1[0] = (int)(3u << 16);                      // data_size = 8B
    d.g1[1] = (int)((units8 & 0xFFFFu) << 16);      // tensor_dim0[15:0] @bits63:48
    d.g1[2] = (int)((units8 >> 16) | (1u << 16));   // tensor_dim0[31:16], tensor_dim1=1
    d.g1[3] = (int)((units8 & 0xFFFFu) << 16);      // tile_dim0 @bits127:112
    d.g1[4] = 1;                                    // tile_dim1 = 1
    d.g1[5] = (int)units8;                          // tensor_dim0_stride lo
    d.g1[6] = 0;
    d.g1[7] = 0;
    return d;
}

#if HAVE_TDM
__device__ __forceinline__ void tdm_load(const TdmDesc& d) {
    i32x4 z4 = {0, 0, 0, 0};
#if __clang_major__ >= 23
    i32x8 z8 = {0, 0, 0, 0, 0, 0, 0, 0};
    __builtin_amdgcn_tensor_load_to_lds(d.g0, d.g1, z4, z4, z8, 0);
#else
    __builtin_amdgcn_tensor_load_to_lds(d.g0, d.g1, z4, z4, 0);
#endif
}
__device__ __forceinline__ void tdm_store(const TdmDesc& d) {
    i32x4 z4 = {0, 0, 0, 0};
#if __clang_major__ >= 23
    i32x8 z8 = {0, 0, 0, 0, 0, 0, 0, 0};
    __builtin_amdgcn_tensor_store_from_lds(d.g0, d.g1, z4, z4, z8, 0);
#else
    __builtin_amdgcn_tensor_store_from_lds(d.g0, d.g1, z4, z4, 0);
#endif
}
#endif

// ---------------------------------------------------------------------------
// Prep: cast + transpose all weights to bf16 in d_ws.
//   ws elements (bf16):
//   [0,16384)        WUt[128][128]  = WU^T
//   [16384,32768)    WVt[128][128]  = WV^T
//   [32768,65536)    W1t[128][256]  = W1^T   (W1 is [256,128])
//   [65536,114688)   W2t[384][128]  = W2^T   (W2 is [128,384])
// ---------------------------------------------------------------------------
__global__ void prep_weights(const float* __restrict__ WU,
                             const float* __restrict__ WV,
                             const float* __restrict__ W1,
                             const float* __restrict__ W2,
                             __hip_bfloat16* __restrict__ ws) {
    int i = blockIdx.x * blockDim.x + threadIdx.x;
    const int stride = gridDim.x * blockDim.x;
    for (; i < 114688; i += stride) {
        float val;
        if (i < 16384) {
            const int g = i >> 7, f = i & 127;
            val = WU[f * 128 + g];
        } else if (i < 32768) {
            const int j = i - 16384;
            const int g = j >> 7, f = j & 127;
            val = WV[f * 128 + g];
        } else if (i < 65536) {
            const int j = i - 32768;
            const int g = j >> 8, f = j & 255;
            val = W1[f * 128 + g];
        } else {
            const int j = i - 65536;
            const int g = j >> 7, f = j & 127;
            val = W2[f * 384 + g];
        }
        ws[i] = __float2bfloat16(val);
    }
}

// ---------------------------------------------------------------------------
// Fused main kernel: 64 rows per block, 8 wave32s. 160 KB static LDS.
// ---------------------------------------------------------------------------
__launch_bounds__(256, 1)
__global__ void painn_update(const float* __restrict__ s,
                             const float* __restrict__ v,
                             const float* __restrict__ b1,
                             const float* __restrict__ b2,
                             const __hip_bfloat16* __restrict__ ws,
                             float* __restrict__ out_ds,
                             float* __restrict__ out_dv) {
    __shared__ alignas(16) __hip_bfloat16 Av[3][TILE_N][FDIM];     // 48 KB
    __shared__ alignas(16) __hip_bfloat16 As[TILE_N][2 * FDIM];    // 32 KB [s || ||V||]
    __shared__ alignas(16) float          UVs[TILE_N][FDIM];       // 32 KB (later: deltasu)
    __shared__ alignas(16) __hip_bfloat16 Hs[TILE_N][FDIM];        // 16 KB
    __shared__ alignas(16) __hip_bfloat16 WUlds[FDIM][FDIM];       // 32 KB (TDM-staged WU^T)

    const int tid  = threadIdx.x;
    const int lane = tid & 31;
    const int wave = tid >> 5;
    const int n0   = blockIdx.x * TILE_N;

    const __hip_bfloat16* WVt = ws + 16384;
    const __hip_bfloat16* W1t = ws + 32768;
    const __hip_bfloat16* W2t = ws + 65536;

    // warm weights into cache hierarchy (global_prefetch_b8)
    if (tid < 128) __builtin_prefetch(ws + tid * 896, 0, 1);

    // ---- TDM: stage WU^T (32 KB bf16) into LDS, overlapped with the -------
    //      f32->bf16 activation staging below.
#if HAVE_TDM
    if (wave == 0) {
        TdmDesc d = tdm_desc_1d((unsigned int)(uintptr_t)&WUlds[0][0], ws, 4096u);
        tdm_load(d);
    }
#else
    for (int i = tid; i < FDIM * FDIM / 4; i += 256) {   // fallback: plain copy
        ((u32x4*)&WUlds[0][0])[i * 0 + i] = ((const u32x4*)ws)[i];
    }
#endif

    // ---- load activation tile, f32 -> bf16, deinterleave v ----------------
    for (int i = tid; i < TILE_N * FDIM; i += 256) {
        const int row = i >> 7, f = i & 127;
        const size_t n = (size_t)(n0 + row);
        const float* vp = v + (n * FDIM + f) * 3;
        Av[0][row][f] = __float2bfloat16(vp[0]);
        Av[1][row][f] = __float2bfloat16(vp[1]);
        Av[2][row][f] = __float2bfloat16(vp[2]);
        As[row][f]    = __float2bfloat16(s[n * FDIM + f]);
    }
#if HAVE_TDM
    if (wave == 0) __builtin_amdgcn_s_wait_tensorcnt(0);
#endif
    __syncthreads();

    // ---- Stage A: U_k = v_k @ WU, V_k = v_k @ WV, fused into --------------
    //      UV = sum_k U_k*V_k, nV = sqrt(sum_k V_k^2) (never materialize U,V)
#pragma unroll
    for (int t = 0; t < 4; ++t) {
        const int tile = wave * 4 + t;
        const int m = tile >> 3, g = tile & 7;
        bf16x16 bu[4], bv[4];
#pragma unroll
        for (int q = 0; q < 4; ++q) {
            bu[q] = ld_frag_b(&WUlds[0][0], FDIM, g * 16, q * 32, lane);  // LDS
            bv[q] = ld_frag_b(WVt, FDIM, g * 16, q * 32, lane);           // L2
        }
        f32x8 uv = {}, nv2 = {};
#pragma unroll
        for (int k = 0; k < 3; ++k) {
            f32x8 au = {}, avv = {};
#pragma unroll
            for (int q = 0; q < 4; ++q) {
                bf16x16 a = ld_frag_a(&Av[k][0][0], FDIM, m * 16, q * 32, lane);
                au  = wmma_bf16(a, bu[q], au);
                avv = wmma_bf16(a, bv[q], avv);
            }
            uv  += au * avv;
            nv2 += avv * avv;
        }
        const int col   = g * 16 + (lane & 15);
        const int rbase = m * 16 + (lane >> 4) * 8;
#pragma unroll
        for (int r = 0; r < 8; ++r) {
            UVs[rbase + r][col]       = uv[r];
            As[rbase + r][FDIM + col] = __float2bfloat16(sqrtf(nv2[r]));
        }
    }
    __syncthreads();

    // ---- Stage B: h = silu([s||nV] @ W1 + b1) -----------------------------
#pragma unroll
    for (int t = 0; t < 4; ++t) {
        const int tile = wave * 4 + t;
        const int m = tile >> 3, g = tile & 7;
        f32x8 acc = {};
#pragma unroll
        for (int q = 0; q < 8; ++q) {
            bf16x16 a = ld_frag_a(&As[0][0], 2 * FDIM, m * 16, q * 32, lane);
            bf16x16 b = ld_frag_b(W1t, 2 * FDIM, g * 16, q * 32, lane);
            acc = wmma_bf16(a, b, acc);
        }
        const int   col   = g * 16 + (lane & 15);
        const float bias  = b1[col];
        const int   rbase = m * 16 + (lane >> 4) * 8;
#pragma unroll
        for (int r = 0; r < 8; ++r) {
            const float x = acc[r] + bias;
            const float y = x * (1.0f / (1.0f + __expf(-x)));  // SiLU
            Hs[rbase + r][col] = __float2bfloat16(y);
        }
    }
    __syncthreads();

    // ---- Stage C: o = h @ W2 + b2; split top/middle/bottom ----------------
    // waves 0..3: "top" tiles -> deltavu = v * top        (direct to global)
    // waves 4..7: middle+bottom -> deltasu = mid*UV + bot (in-place into UVs,
    //             then one TDM tensor_store_from_lds streams the 32 KB tile)
#pragma unroll
    for (int t = 0; t < 8; ++t) {
        const int u = wave * 8 + t;
        if (u < 32) {
            const int m = u >> 3, g = u & 7;
            f32x8 acc = {};
#pragma unroll
            for (int q = 0; q < 4; ++q) {
                bf16x16 a = ld_frag_a(&Hs[0][0], FDIM, m * 16, q * 32, lane);
                bf16x16 b = ld_frag_b(W2t, FDIM, g * 16, q * 32, lane);
                acc = wmma_bf16(a, b, acc);
            }
            const int   col   = g * 16 + (lane & 15);
            const float bias  = b2[col];
            const int   rbase = m * 16 + (lane >> 4) * 8;
#pragma unroll
            for (int r = 0; r < 8; ++r) {
                const int row = rbase + r;
                const float top = acc[r] + bias;
                const size_t n = (size_t)(n0 + row);
                float* o = out_dv + (n * FDIM + col) * 3;
                o[0] = __bfloat162float(Av[0][row][col]) * top;
                o[1] = __bfloat162float(Av[1][row][col]) * top;
                o[2] = __bfloat162float(Av[2][row][col]) * top;
            }
        } else {
            const int u2 = u - 32;
            const int m = u2 >> 3, g = u2 & 7;
            f32x8 accM = {}, accB = {};
#pragma unroll
            for (int q = 0; q < 4; ++q) {
                bf16x16 a  = ld_frag_a(&Hs[0][0], FDIM, m * 16, q * 32, lane);
                bf16x16 bm = ld_frag_b(W2t, FDIM, FDIM + g * 16, q * 32, lane);
                bf16x16 bb = ld_frag_b(W2t, FDIM, 2 * FDIM + g * 16, q * 32, lane);
                accM = wmma_bf16(a, bm, accM);
                accB = wmma_bf16(a, bb, accB);
            }
            const int   col   = g * 16 + (lane & 15);
            const float biasM = b2[FDIM + col];
            const float biasB = b2[2 * FDIM + col];
            const int   rbase = m * 16 + (lane >> 4) * 8;
#pragma unroll
            for (int r = 0; r < 8; ++r) {
                const int row = rbase + r;
                UVs[row][col] =
                    (accM[r] + biasM) * UVs[row][col] + (accB[r] + biasB);
            }
        }
    }
    __syncthreads();

    // ---- stream deltasu tile (contiguous 32 KB) out via TDM ---------------
#if HAVE_TDM
    if (wave == 0) {
        TdmDesc d = tdm_desc_1d((unsigned int)(uintptr_t)&UVs[0][0],
                                out_ds + (size_t)n0 * FDIM, 4096u);
        tdm_store(d);
        // S_ENDPGM performs an implicit wait-idle which drains TENSORcnt.
    }
#else
    for (int i = tid; i < TILE_N * FDIM; i += 256) {
        const int row = i >> 7, f = i & 127;
        out_ds[(size_t)(n0 + row) * FDIM + f] = UVs[row][f];
    }
#endif
}

// ---------------------------------------------------------------------------
extern "C" void kernel_launch(void* const* d_in, const int* in_sizes, int n_in,
                              void* d_out, int out_size, void* d_ws, size_t ws_size,
                              hipStream_t stream) {
    (void)in_sizes; (void)n_in; (void)out_size; (void)ws_size;
    const float* s  = (const float*)d_in[0];
    const float* v  = (const float*)d_in[1];
    const float* WU = (const float*)d_in[2];
    const float* WV = (const float*)d_in[3];
    const float* W1 = (const float*)d_in[4];
    const float* b1 = (const float*)d_in[5];
    const float* W2 = (const float*)d_in[6];
    const float* b2 = (const float*)d_in[7];

    __hip_bfloat16* ws = (__hip_bfloat16*)d_ws;
    float* out_ds = (float*)d_out;
    float* out_dv = out_ds + (size_t)NROWS * FDIM;

    prep_weights<<<112, 256, 0, stream>>>(WU, WV, W1, W2, ws);
    painn_update<<<NROWS / TILE_N, 256, 0, stream>>>(s, v, b1, b2, ws,
                                                     out_ds, out_dv);
}
